// LSTM_64441689309589
// MI455X (gfx1250) — matrix operand order repaired and checked by reading, not compiled, not run on hardware
//
#include <hip/hip_runtime.h>
#include <hip/hip_bf16.h>

typedef __attribute__((ext_vector_type(16))) _Float16 v16h;
typedef __attribute__((ext_vector_type(8)))  _Float16 v8h;
typedef __attribute__((ext_vector_type(8)))  float    v8f;
typedef __attribute__((ext_vector_type(2)))  float    v2f;

#define VOCAB 50257
#define EMB   256
#define HID   512
#define BATCH 64
#define SEQ   512
#define NCLS  50257
#define NWG   32           // persistent workgroups for the recurrence
#define KTOT  (EMB + HID)  // 768
#define WPAD  776          // LDS row stride in halves (768 + 8): 388 DW == 4 banks mod 64

__device__ __forceinline__ float sigf(float x)  { return 1.0f / (1.0f + __expf(-x)); }
__device__ __forceinline__ float tanhf_(float x){ return 2.0f / (1.0f + __expf(-2.0f * x)) - 1.0f; }

// ---------------- prep: embedding gather into (t, b, k) f16 ----------------
__global__ void gather_emb(const int* __restrict__ x, const float* __restrict__ emb,
                           _Float16* __restrict__ X16) {
  int i = blockIdx.x * blockDim.x + threadIdx.x;   // SEQ*BATCH*EMB total, exact grid
  int k = i & (EMB - 1);
  int b = (i >> 8) & (BATCH - 1);
  int t = i >> 14;
  int tok = x[b * SEQ + t];
  X16[i] = (_Float16)emb[(size_t)tok * EMB + k];
}

// ---------------- persistent LSTM recurrence ----------------
// 32 WGs x 256 threads. WG wg owns h columns [wg*16, wg*16+16).
// wave = (m, gp): m = M-tile (rows m*16..m*16+15), gp=0 -> gates f,i ; gp=1 -> gates o,c~.
// Weight slice (64 gate-rows x 768) staged once into LDS as f16, padded rows.
__global__ void lstm_kernel(const float*    __restrict__ W,     // (2048, 768) fp32
                            const _Float16* __restrict__ X16,   // (SEQ, 64, 256) f16
                            const float*    __restrict__ bias,  // (2048,)
                            _Float16*                    h16,   // (64, 512) f16, shared via L2
                            unsigned*                    cnt)   // grid barrier counter (pre-zeroed)
{
  extern __shared__ char smem[];
  _Float16* sW = (_Float16*)smem;                        // 64 rows x WPAD halves (~97 KB)
  float*    sF = (float*)(smem + 64 * WPAD * 2);         // [4][8][32]
  float*    sI = sF + 4 * 8 * 32;                        // [4][8][32]

  const int lane  = threadIdx.x & 31;
  const int wave  = threadIdx.x >> 5;
  const int m     = wave & 3;
  const int gp    = wave >> 2;
  const int n16   = lane & 15;
  const int hi    = lane >> 4;       // 0 for lanes 0-15, 1 for lanes 16-31
  const int cbase = blockIdx.x * 16;

  // ---- stage + convert weight slice into LDS (once) ----
  for (int idx = threadIdx.x; idx < 64 * (KTOT / 8); idx += 256) {
    int r = idx / (KTOT / 8);            // local row 0..63  (gate g = r>>4, col rr = r&15)
    int c = idx % (KTOT / 8);            // 8-float chunk
    int gRow = (r >> 4) * HID + cbase + (r & 15);
    const float* src = W + (size_t)gRow * KTOT + c * 8;
    v8h d;
    #pragma unroll
    for (int j = 0; j < 8; ++j) d[j] = (_Float16)src[j];
    *(v8h*)(sW + r * WPAD + c * 8) = d;
  }
  __syncthreads();

  const int rowA = m * 16 + n16;                  // A-matrix M index for this lane
  const int g0   = gp * 2;                        // gate index pair
  const int r0   = (g0 + 0) * 16 + n16;           // LDS-local weight rows
  const int r1   = (g0 + 1) * 16 + n16;
  const float bias0 = bias[(g0 + 0) * HID + cbase + n16];
  const float bias1 = bias[(g0 + 1) * HID + cbase + n16];

  v8f C = {};   // cell-state tile (meaningful in gp==1 waves, lives in VGPRs all 512 steps)

  unsigned wofs = 0;  // always 0; made opaque each step to defeat LICM (prevents VGPR spills)

  for (int t = 0; t < SEQ; ++t) {
    asm volatile("" : "+s"(wofs));     // compiler must treat LDS weight base as loop-variant
    const _Float16* sW0 = sW + wofs + (size_t)r0 * WPAD;
    const _Float16* sW1 = sW + wofs + (size_t)r1 * WPAD;

    v8f acc0 = {};
    v8f acc1 = {};
    const _Float16* Xt = X16 + (size_t)t * BATCH * EMB;

    #pragma unroll
    for (int kk = 0; kk < KTOT / 32; ++kk) {       // 24 k-steps of K=32
      const int k0 = kk * 32;
      // ---- A tile (16x32 f16): per-lane two contiguous 8-half chunks from global ----
      const _Float16* abase;
      int astride, kloc;
      if (k0 < EMB) { abase = Xt;  astride = EMB; kloc = k0; }
      else          { abase = h16; astride = HID; kloc = k0 - EMB; }
      const _Float16* ap = abase + (size_t)rowA * astride + kloc + hi * 8;
      v8h alo = *(const v8h*)(ap);
      v8h ahi = *(const v8h*)(ap + 16);
      v16h a = __builtin_shufflevector(alo, ahi, 0,1,2,3,4,5,6,7,8,9,10,11,12,13,14,15);

      // ---- B tiles (32x16 f16): lane reads two 8-half chunks of its weight row (LDS), same k order as A ----
      const _Float16* b0p = sW0 + k0 + hi * 8;
      const _Float16* b1p = sW1 + k0 + hi * 8;
      v8h b0lo = *(const v8h*)(b0p);
      v8h b0hi = *(const v8h*)(b0p + 16);
      v8h b1lo = *(const v8h*)(b1p);
      v8h b1hi = *(const v8h*)(b1p + 16);
      v16h b0 = __builtin_shufflevector(b0lo, b0hi, 0,1,2,3,4,5,6,7,8,9,10,11,12,13,14,15);
      v16h b1 = __builtin_shufflevector(b1lo, b1hi, 0,1,2,3,4,5,6,7,8,9,10,11,12,13,14,15);

      acc0 = __builtin_amdgcn_wmma_f32_16x16x32_f16(false, a, false, b0, (short)0, acc0, false, false);
      acc1 = __builtin_amdgcn_wmma_f32_16x16x32_f16(false, a, false, b1, (short)0, acc1, false, false);
    }

    #pragma unroll
    for (int v = 0; v < 8; ++v) { acc0[v] += bias0; acc1[v] += bias1; }

    if (gp == 0) {  // forget & input gates -> LDS
      #pragma unroll
      for (int v = 0; v < 8; ++v) {
        sF[(m * 8 + v) * 32 + lane] = sigf(acc0[v]);
        sI[(m * 8 + v) * 32 + lane] = sigf(acc1[v]);
      }
    }
    __syncthreads();
    if (gp == 1) {  // output gate & candidate; update C (registers), write h16 slice
      #pragma unroll
      for (int v = 0; v < 8; ++v) {
        float o  = sigf(acc0[v]);
        float ct = tanhf_(acc1[v]);
        float Cn = sF[(m * 8 + v) * 32 + lane] * C[v] + sI[(m * 8 + v) * 32 + lane] * ct;
        C[v] = Cn;
        float h  = o * tanhf_(Cn);
        int brow = m * 16 + v + hi * 8;
        h16[(size_t)brow * HID + cbase + n16] = (_Float16)h;
      }
    }
    __threadfence();     // release h16 stores to device scope
    __syncthreads();     // WG done (also guards sF/sI reuse)

    if (t + 1 < SEQ) {   // grid-wide barrier: all WGs' h16 visible before next step
      if (threadIdx.x == 0) {
        atomicAdd(cnt, 1u);
        unsigned tgt = (unsigned)NWG * (unsigned)(t + 1);
        while (__hip_atomic_load(cnt, __ATOMIC_RELAXED, __HIP_MEMORY_SCOPE_AGENT) < tgt)
          __builtin_amdgcn_s_sleep(1);
      }
      __syncthreads();
      __threadfence();   // acquire: next-step h16 loads see fresh data
    }
  }
}

// ---------------- output FC: out = h @ fc_w^T + fc_b  (bandwidth-bound, f32 WMMA) ----------------
// One 16x16 out tile per wave; 4 M-tiles x 3142 N-tiles = 12568 tiles = 1571 blocks x 8 waves.
__global__ void fc_kernel(const _Float16* __restrict__ h16,   // (64, 512) f16
                          const float*    __restrict__ fcw,   // (50257, 512) fp32
                          const float*    __restrict__ fcb,   // (50257,)
                          float*          __restrict__ out)   // (64, 50257)
{
  const int lane = threadIdx.x & 31;
  const int wave = threadIdx.x >> 5;
  const int hi   = lane >> 4;
  const int n16  = lane & 15;

  const int tile = blockIdx.x * 8 + wave;
  const int mt   = tile & 3;
  const int nt   = tile >> 2;

  const int ncol = nt * 16 + n16;
  const int nrow = (ncol < NCLS) ? ncol : (NCLS - 1);   // clamp for B loads (store masked later)
  const int rowA = mt * 16 + n16;

  v8f acc = {};
  for (int k0 = 0; k0 < HID; k0 += 4) {
    const int ka = k0 + hi * 2;
    v2f a;
    a.x = (float)h16[rowA * HID + ka];
    a.y = (float)h16[rowA * HID + ka + 1];
    const float2 bb = *(const float2*)(fcw + (size_t)nrow * HID + ka);
    v2f b; b.x = bb.x; b.y = bb.y;
    acc = __builtin_amdgcn_wmma_f32_16x16x4_f32(false, a, false, b, (short)0, acc, false, false);
  }
  if (ncol < NCLS) {
    const float bv = fcb[ncol];
    #pragma unroll
    for (int v = 0; v < 8; ++v) {
      int brow = mt * 16 + v + hi * 8;
      out[(size_t)brow * NCLS + ncol] = acc[v] + bv;
    }
  }
}

extern "C" void kernel_launch(void* const* d_in, const int* in_sizes, int n_in,
                              void* d_out, int out_size, void* d_ws, size_t ws_size,
                              hipStream_t stream) {
  const int*   x   = (const int*)d_in[0];
  const float* emb = (const float*)d_in[1];
  const float* W   = (const float*)d_in[2];
  const float* Wb  = (const float*)d_in[3];
  const float* fcw = (const float*)d_in[4];
  const float* fcb = (const float*)d_in[5];
  float* out = (float*)d_out;

  // workspace layout (16B aligned): X16 | h16 | counter
  char* ws = (char*)d_ws;
  const size_t X16_bytes = (size_t)SEQ * BATCH * EMB * 2;           // 16.8 MB
  const size_t H16_bytes = (size_t)BATCH * HID * 2;                 // 64 KB
  _Float16* X16 = (_Float16*)(ws);
  _Float16* h16 = (_Float16*)(ws + X16_bytes);
  unsigned* cnt = (unsigned*)(ws + X16_bytes + H16_bytes);

  gather_emb<<<(SEQ * BATCH * EMB) / 256, 256, 0, stream>>>(x, emb, X16);
  hipMemsetAsync(h16, 0, H16_bytes, stream);        // h0 = 0 (f16 zero is bit-zero)
  hipMemsetAsync(cnt, 0, sizeof(unsigned), stream); // barrier counter reset per launch

  const size_t lds_bytes = (size_t)64 * WPAD * 2 + 2 * 4 * 8 * 32 * sizeof(float); // ~105 KB
  lstm_kernel<<<NWG, 256, lds_bytes, stream>>>(W, X16, Wb, h16, cnt);
  fc_kernel<<<1571, 256, 0, stream>>>(h16, fcw, fcb, out);
}
